// BernsteinFlowModel_2860448219352
// MI455X (gfx1250) — matrix operand order repaired and checked by reading, not compiled, not run on hardware
//
#include <hip/hip_runtime.h>

// ---------------------------------------------------------------------------
// CDNA5 / gfx1250 Bernstein-flow density kernel.
//   - dominant GEMM [8192 x 16384] x [16384 x 32] mapped to
//     v_wmma_f32_16x16x32_bf16 (f32 accumulate), A-tiles synthesized on the
//     fly from the Kronecker (rank-1) structure of the conditioner basis.
//   - i>=6 A-tiles built with v_pk_mul_bf16 (forced via inline asm) from
//     bf16-resident LDS factors: 3 DS loads + 8 packed muls per 2 WMMAs.
//   - B-matrix (batch-independent sigmoid/cumsum/softplus of A) precomputed
//     once per launch into d_ws in WMMA-fragment lane order.
//   - cross-lane reductions via ds_swizzle XOR patterns.
//   - 311 KB dynamic LDS per workgroup (needs CDNA5's 320 KB WGP LDS).
// ---------------------------------------------------------------------------

typedef __attribute__((ext_vector_type(16))) __bf16 v16bf;
typedef __attribute__((ext_vector_type(8)))  __bf16 bf16x8;
typedef __attribute__((ext_vector_type(8)))  float  v8f;
typedef __attribute__((ext_vector_type(4)))  float  f32x4;
typedef __attribute__((ext_vector_type(2)))  float  f32x2;

#define WAVES           8
#define RPW             16                 // batch rows per wave (WMMA M)
#define ROWS_PER_BLOCK  (WAVES * RPW)      // 128
#define NCHUNK_TOTAL    685                // sum over i of max(1, 4^i/32)
#define FRAG_HALVES     (NCHUNK_TOTAL * 1024)
#define FRAG_WORDS     (FRAG_HALVES / 2)   // 350720 u32

__host__ __device__ constexpr int chunkBase(int i) {
  return (i==0)?0:(i==1)?1:(i==2)?2:(i==3)?3:(i==4)?5:(i==5)?13:(i==6)?45:173;
}

__constant__ float BIN31[32] = {
  1.f, 31.f, 465.f, 4495.f, 31465.f, 169911.f, 736281.f, 2629575.f,
  7888725.f, 20160075.f, 44352165.f, 84672315.f, 141120525.f, 206253075.f,
  265182525.f, 300540195.f, 300540195.f, 265182525.f, 206253075.f,
  141120525.f, 84672315.f, 44352165.f, 20160075.f, 7888725.f, 2629575.f,
  736281.f, 169911.f, 31465.f, 4495.f, 465.f, 31.f, 1.f };

// LDS layout (per block):
//   uF  : [WAVES][16][84]  f32   (U1 @0, U2 @4, U3 @20 per row)
//   vF  : [WAVES][16][84]  f32   (V5 @0, V6 @4, V7 @20 per row)
//   eF  : [WAVES][16][8][16][2] f32  (paired: [n0][{E[n0], E[n0+16]}])
//   d31 : [WAVES][16][8]   f32
//   u4  : [WAVES][16][256] bf16  (U4 = P0⊗P1⊗P2⊗P3)
//   vB  : [WAVES][16][96]  bf16  (V7b @0 (64), V6b @64 (16), pad)
#define UF_FLOATS  (WAVES * RPW * 84)
#define EF_FLOATS  (WAVES * RPW * 8 * 32)
#define D31_FLOATS (WAVES * RPW * 8)
#define U4_HALVES  (WAVES * RPW * 256)
#define VB_HALVES  (WAVES * RPW * 96)
#define SMEM_BYTES ((2 * UF_FLOATS + EF_FLOATS + D31_FLOATS) * 4 + (U4_HALVES + VB_HALVES) * 2)

// scale 8 bf16 lanes by one bf16 value using packed bf16 multiplies
// (V_PK_MUL_BF16, VOP3P op 42 — forced via inline asm so clang cannot
// legalize the bf16 vector fmul through f32).
__device__ __forceinline__ bf16x8 pk_scale8(__bf16 s, bf16x8 y) {
  union { __bf16 h; unsigned short u; } c; c.h = s;
  unsigned int sw = ((unsigned int)c.u << 16) | (unsigned int)c.u;  // splat
  union { bf16x8 v; unsigned int w[4]; } b, r;
  b.v = y;
#pragma unroll
  for (int j = 0; j < 4; ++j)
    asm("v_pk_mul_bf16 %0, %1, %2" : "=v"(r.w[j]) : "v"(sw), "v"(b.w[j]));
  return r.v;
}

// wave32 XOR-butterfly reduction within 16-lane half-groups, one DS op per
// level (ds_swizzle group-of-32 encoding: [14:10]=xor, [9:5]=or, [4:0]=and).
__device__ __forceinline__ float xor_reduce16(float t) {
  union { float f; int i; } u;
  u.f = t; u.i = __builtin_amdgcn_ds_swizzle(u.i, 0x041F); t += u.f;  // ^1
  u.f = t; u.i = __builtin_amdgcn_ds_swizzle(u.i, 0x081F); t += u.f;  // ^2
  u.f = t; u.i = __builtin_amdgcn_ds_swizzle(u.i, 0x101F); t += u.f;  // ^4
  u.f = t; u.i = __builtin_amdgcn_ds_swizzle(u.i, 0x201F); t += u.f;  // ^8
  return t;
}

// ---------------------------------------------------------------------------
// prep kernels
// ---------------------------------------------------------------------------
__global__ void zfrag_kernel(unsigned int* w, int n) {
  int t = blockIdx.x * 256 + threadIdx.x;
  if (t < n) w[t] = 0u;
}

// c_i[n, m] = sigmoid(cumsum_n softplus(A[i, n, m])), written in B-fragment
// lane order: frag[((chunkBase(i)+m/32)*2 + n/16)*512 + lane*16 + e]
// with lane = (n&15) + ((m&8) ? 16 : 0), e = (m&7) + ((m&31)>=16 ? 8 : 0).  Row n=31 and
// m >= 4^i stay zero (zfrag), which also zeroes padded K for i<3.
__global__ void cfill_kernel(const float* __restrict__ A, __bf16* __restrict__ frag) {
  int tid = blockIdx.x * 256 + threadIdx.x;     // 8 * 16384 threads
  int i = tid >> 14;
  int m = tid & 16383;
  int K = 1 << (2 * i);
  if (m >= K) return;
  int chunk = m >> 5, moff = m & 31;
  int laneAdd = (moff & 8) ? 16 : 0;
  int e = (moff & 7) + ((moff >= 16) ? 8 : 0);
  size_t base = (size_t)(chunkBase(i) + chunk) * 2;
  const float* Ai = A + ((size_t)i * 31) * 16384 + m;
  float s = 0.f;
  for (int n = 0; n < 31; ++n) {
    s += log1pf(__expf(Ai[(size_t)n * 16384]));      // softplus, a in [0,1]
    float cv = 1.f / (1.f + __expf(-s));             // sigmoid
    int h = n >> 4, lane = (n & 15) + laneAdd;
    frag[(base + h) * 512 + lane * 16 + e] = (__bf16)cv;
  }
}

// ---------------------------------------------------------------------------
// main kernel: one wave = one 16-row batch tile
// ---------------------------------------------------------------------------
template <int I>
__device__ __forceinline__ void step_i(const float* __restrict__ uFw,
                                       const __bf16* __restrict__ u4w,
                                       const float* __restrict__ vFw,
                                       const __bf16* __restrict__ vBw,
                                       const float* __restrict__ eFw,
                                       const float* __restrict__ d31w,
                                       const v16bf* __restrict__ fragV,
                                       int lane, float& dens) {
  constexpr int K4   = 1 << (2 * I);
  constexpr int NCH  = (K4 > 32) ? (K4 / 32) : 1;
  constexpr int CB   = chunkBase(I);
  constexpr int UOFF = (I == 1) ? 0 : (I == 2) ? 4 : 20;

  const int Mrow = lane & 15;
  const int kb   = (lane < 16) ? 0 : 8;     // A-fragment K base (ISA layout)
  const float*  urow  = uFw + Mrow * 84 + UOFF;
  const float*  vrow  = vFw + Mrow * 84;    // V5 @0 (f32, used by I==5)
  const __bf16* u4row = u4w + Mrow * 256;
  const __bf16* vbrow = vBw + Mrow * 96;    // V7b @0, V6b @64 (bf16)

  v8f acc0 = {0.f,0.f,0.f,0.f,0.f,0.f,0.f,0.f};
  v8f acc1 = acc0;

  for (int c = 0; c < NCH; ++c) {
    // ---- synthesize A fragment: C_I[m] = U[m>>s] * V[m & mask] ----
    v16bf a;
    if constexpr (I == 4) {
      // A tile is U4 itself: pure register concatenation, no conversions.
      bf16x8 q0 = *(const bf16x8*)(u4row + c * 32 + kb);
      bf16x8 q1 = *(const bf16x8*)(u4row + c * 32 + kb + 16);
      a = __builtin_shufflevector(q0, q1, 0,1,2,3,4,5,6,7,8,9,10,11,12,13,14,15);
    } else if constexpr (I == 6) {
      // val(e) = (e<8 ? U4[2c] : U4[2c+1]) * V6b[kb + (e&7)]
      bf16x8 g  = *(const bf16x8*)(vbrow + 64 + kb);
      bf16x8 lo = pk_scale8(u4row[2 * c],     g);   // v_pk_mul_bf16
      bf16x8 hi = pk_scale8(u4row[2 * c + 1], g);
      a = __builtin_shufflevector(lo, hi, 0,1,2,3,4,5,6,7,8,9,10,11,12,13,14,15);
    } else if constexpr (I == 7) {
      // U index constant within a 32-K chunk; V groups {+0..7} and {+16..23}
      const __bf16* vg = vbrow + ((c & 1) << 5) + kb;
      bf16x8 g0 = *(const bf16x8*)(vg);
      bf16x8 g1 = *(const bf16x8*)(vg + 16);
      __bf16 uf = u4row[c >> 1];
      bf16x8 lo = pk_scale8(uf, g0);                // v_pk_mul_bf16
      bf16x8 hi = pk_scale8(uf, g1);
      a = __builtin_shufflevector(lo, hi, 0,1,2,3,4,5,6,7,8,9,10,11,12,13,14,15);
    } else {
      f32x4 v0, v1, v2, v3;
      f32x4 vv4;
      float uu[4];
      if constexpr (I == 3) {
        const f32x4* g = (const f32x4*)(urow + c * 32 + kb);
        v0 = g[0]; v1 = g[1]; v2 = g[4]; v3 = g[5];
      } else if constexpr (I == 5) {
        vv4 = *(const f32x4*)(vrow);
        int ub = 8 * c + (kb >> 2);
        uu[0] = (float)u4row[ub + 0];
        uu[1] = (float)u4row[ub + 1];
        uu[2] = (float)u4row[ub + 4];
        uu[3] = (float)u4row[ub + 5];
      }
#pragma unroll
      for (int e = 0; e < 16; ++e) {
        // ISA A-layout: lane<16 K={0..7,16..23}; lane>=16 K={8..15,24..31}
        const int Ke = ((e < 8) ? e : e + 8) + kb;
        float val;
        if constexpr (I == 0) {
          val = (Ke == 0) ? 1.f : 0.f;
        } else if constexpr (I <= 2) {
          float t = urow[Ke];                      // in-row LDS, defined data
          val = (Ke < K4) ? t : 0.f;               // pad K beyond 4^I
        } else if constexpr (I == 3) {
          val = (e < 4) ? v0[e] : (e < 8) ? v1[e - 4]
              : (e < 12) ? v2[e - 8] : v3[e - 12];
        } else {  // I == 5
          val = uu[e >> 2] * vv4[e & 3];
        }
        a[e] = (__bf16)val;
      }
    }

    // ---- B fragments (pre-swizzled bf16 in L2-resident workspace) ----
    const v16bf* bp = fragV + (size_t)(CB + c) * 2 * 32 + lane;
    v16bf b0 = bp[0];       // columns n = 0..15
    v16bf b1 = bp[32];      // columns n = 16..31
    if (c + 1 < NCH)
      __builtin_prefetch(fragV + (size_t)(CB + c + 1) * 2 * 32 + lane, 0, 1);

    acc0 = __builtin_amdgcn_wmma_f32_16x16x32_bf16(false, a, false, b0,
                                                   (short)0, acc0, false, false);
    acc1 = __builtin_amdgcn_wmma_f32_16x16x32_bf16(false, a, false, b1,
                                                   (short)0, acc1, false, false);
  }

  // ---- factor_I[row] = sum_n betas[row,n] * E[row,I,n] + D31[row,I] ----
  // C/D layout: VGPR v, lanes 0-15 -> M=v, N=lane; lanes 16-31 -> M=v+8.
  const int n0 = lane & 15;
#pragma unroll
  for (int v = 0; v < 8; ++v) {
    int M = v + ((lane < 16) ? 0 : 8);
    f32x2 ee = *((const f32x2*)(eFw + (M * 8 + I) * 32) + n0);  // ds_load_b64
    float t = acc0[v] * ee.x + acc1[v] * ee.y;
    t = xor_reduce16(t);
    float f = t + d31w[M * 8 + I];
    dens *= ((lane & 15) == v) ? f : 1.f;   // branchless owner update
  }
}

__global__ void __launch_bounds__(256, 1)
flow_main_kernel(const float* __restrict__ x, const __bf16* __restrict__ frag,
                 float* __restrict__ out) {
  const int lane = threadIdx.x & 31;
  const int wave = threadIdx.x >> 5;
  const int rowBase = (blockIdx.x * WAVES + wave) * RPW;

  extern __shared__ char smem[];
  float*  uF  = (float*)smem;
  float*  vF  = uF + UF_FLOATS;
  float*  eF  = vF + UF_FLOATS;
  float*  d31 = eF + EF_FLOATS;
  __bf16* u4  = (__bf16*)(d31 + D31_FLOATS);
  __bf16* vB  = u4 + U4_HALVES;

  float*  uFw  = uF  + wave * RPW * 84;
  float*  vFw  = vF  + wave * RPW * 84;
  float*  eFw  = eF  + wave * RPW * 8 * 32;
  float*  d31w = d31 + wave * RPW * 8;
  __bf16* u4w  = u4  + wave * RPW * 256;
  __bf16* vBw  = vB  + wave * RPW * 96;

  auto bern3 = [](float t, float* p) {
    float o = 1.f - t;
    p[0] = o * o * o; p[1] = 3.f * t * o * o;
    p[2] = 3.f * t * t * o; p[3] = t * t * t;
  };

  // ---------------- phase 0: per-row rank-1 factors + E basis ----------------
  {
    const int r = rowBase + (lane & 15);
    if (lane < 16) {
      float P0[4], P1[4], P2[4], P3[4];
      bern3(x[r * 8 + 0], P0); bern3(x[r * 8 + 1], P1);
      bern3(x[r * 8 + 2], P2); bern3(x[r * 8 + 3], P3);
      float* u = uFw + lane * 84;
#pragma unroll
      for (int a0 = 0; a0 < 4; ++a0) u[a0] = P0[a0];
#pragma unroll
      for (int a0 = 0; a0 < 4; ++a0)
#pragma unroll
        for (int b0 = 0; b0 < 4; ++b0) u[4 + a0 * 4 + b0] = P0[a0] * P1[b0];
      for (int a0 = 0; a0 < 16; ++a0)
#pragma unroll
        for (int b0 = 0; b0 < 4; ++b0) u[20 + a0 * 4 + b0] = u[4 + a0] * P2[b0];
      __bf16* q = u4w + lane * 256;
      for (int a0 = 0; a0 < 64; ++a0) {
        float t = u[20 + a0];
#pragma unroll
        for (int b0 = 0; b0 < 4; ++b0) q[a0 * 4 + b0] = (__bf16)(t * P3[b0]);
      }
    } else {
      float Q0[4], Q1[4], Q2[4];
      bern3(x[r * 8 + 4], Q0); bern3(x[r * 8 + 5], Q1); bern3(x[r * 8 + 6], Q2);
      float* vv = vFw + (lane & 15) * 84;
      __bf16* vb = vBw + (lane & 15) * 96;
#pragma unroll
      for (int a0 = 0; a0 < 4; ++a0) vv[a0] = Q0[a0];
#pragma unroll
      for (int a0 = 0; a0 < 4; ++a0)
#pragma unroll
        for (int b0 = 0; b0 < 4; ++b0) vv[4 + a0 * 4 + b0] = Q0[a0] * Q1[b0];
      for (int a0 = 0; a0 < 16; ++a0)
#pragma unroll
        for (int b0 = 0; b0 < 4; ++b0) vv[20 + a0 * 4 + b0] = vv[4 + a0] * Q2[b0];
      // bf16 mirrors: V7b @0 (64), V6b @64 (16)
      for (int a0 = 0; a0 < 64; ++a0) vb[a0] = (__bf16)vv[20 + a0];
#pragma unroll
      for (int a0 = 0; a0 < 16; ++a0) vb[64 + a0] = (__bf16)vv[4 + a0];
    }
  }
  // E basis, paired layout: eF[row][i][n0] = {D[n0]-D[n0+1], D[n0+16]-D[n0+17]}
  // (E[31] = 0), D[k] = 32*C(31,k)*x^k*(1-x)^(31-k), computed in registers.
  for (int t4 = 0; t4 < 4; ++t4) {
    int p = lane + 32 * t4;            // 128 (row,i) pairs per wave
    int row = p >> 3, i = p & 7;
    float xv = x[(size_t)(rowBase + row) * 8 + i];
    float dk[32];
    float xp = 1.f;
#pragma unroll
    for (int k = 0; k < 32; ++k) { dk[k] = xp; xp *= xv; }
    float om = 1.f - xv, g = 1.f;
#pragma unroll
    for (int k = 31; k >= 0; --k) { dk[k] *= 32.f * BIN31[k] * g; g *= om; }
    d31w[row * 8 + i] = dk[31];
    float* e = eFw + (row * 8 + i) * 32;
#pragma unroll
    for (int n = 0; n < 16; ++n) {
      e[2 * n]     = dk[n] - dk[n + 1];
      e[2 * n + 1] = (n + 16 < 31) ? (dk[n + 16] - dk[n + 17]) : 0.f;
    }
  }
  __syncthreads();

  // ---------------- phase 1: eight WMMA contractions ----------------
  const v16bf* fragV = (const v16bf*)frag;
  float dens = 1.f;
  step_i<0>(uFw, u4w, vFw, vBw, eFw, d31w, fragV, lane, dens);
  step_i<1>(uFw, u4w, vFw, vBw, eFw, d31w, fragV, lane, dens);
  step_i<2>(uFw, u4w, vFw, vBw, eFw, d31w, fragV, lane, dens);
  step_i<3>(uFw, u4w, vFw, vBw, eFw, d31w, fragV, lane, dens);
  step_i<4>(uFw, u4w, vFw, vBw, eFw, d31w, fragV, lane, dens);
  step_i<5>(uFw, u4w, vFw, vBw, eFw, d31w, fragV, lane, dens);
  step_i<6>(uFw, u4w, vFw, vBw, eFw, d31w, fragV, lane, dens);
  step_i<7>(uFw, u4w, vFw, vBw, eFw, d31w, fragV, lane, dens);

  if ((lane & 15) < 8) {
    int row = (lane & 15) + ((lane < 16) ? 0 : 8);
    out[rowBase + row] = dens;
  }
}

// ---------------------------------------------------------------------------
extern "C" void kernel_launch(void* const* d_in, const int* in_sizes, int n_in,
                              void* d_out, int out_size, void* d_ws, size_t ws_size,
                              hipStream_t stream) {
  (void)in_sizes; (void)n_in; (void)ws_size;
  const float* x = (const float*)d_in[0];        // [8192, 8]
  const float* A = (const float*)d_in[1];        // [8, 31, 16384]
  float* out = (float*)d_out;                    // [8192]
  __bf16* frag = (__bf16*)d_ws;                  // 1.37 MB of B fragments

  zfrag_kernel<<<(FRAG_WORDS + 255) / 256, 256, 0, stream>>>((unsigned int*)d_ws,
                                                             FRAG_WORDS);
  cfill_kernel<<<(8 * 16384) / 256, 256, 0, stream>>>(A, frag);

  (void)hipFuncSetAttribute((const void*)flow_main_kernel,
                            hipFuncAttributeMaxDynamicSharedMemorySize,
                            SMEM_BYTES);
  int grid = out_size / ROWS_PER_BLOCK;          // 8192 / 128 = 64
  flow_main_kernel<<<grid, 256, SMEM_BYTES, stream>>>(x, frag, out);
}
